// GAT_40810779246848
// MI455X (gfx1250) — matrix operand order repaired and checked by reading, not compiled, not run on hardware
//
#include <hip/hip_runtime.h>
#include <stdint.h>

// ---------------------------------------------------------------------------
// Types for CDNA5 WMMA (wave32): D[16x16 f32] = A[16x32 f16] * B[32x16 f16] + C
// ---------------------------------------------------------------------------
typedef __attribute__((ext_vector_type(16))) _Float16 v16h;
typedef __attribute__((ext_vector_type(8)))  float    v8f;

union AFrag { v16h v; _Float16 h[16]; };

__device__ __forceinline__ void atomicAddF(float* p, float v) {
  // relaxed, agent scope -> native global_atomic_add_f32 in L2
  __hip_atomic_fetch_add(p, v, __ATOMIC_RELAXED, __HIP_MEMORY_SCOPE_AGENT);
}

// ---------------------------------------------------------------------------
// generic fill
// ---------------------------------------------------------------------------
__global__ void fill_u32_kernel(unsigned* __restrict__ p, unsigned v, long long n) {
  long long i = (long long)blockIdx.x * blockDim.x + threadIdx.x;
  if (i < n) p[i] = v;
}

// ---------------------------------------------------------------------------
// Pack W [K,256] fp32 row-major -> per-lane contiguous f16 B-fragments.
// Linear layout: P[(((kt*16 + ct)*32 + lane)*16) + j]
//   lane<16 : col = ct*16+lane,    k = kt*32 + j        (j=0..15)
//   lane>=16: col = ct*16+lane-16, k = kt*32 + 16 + j
// ---------------------------------------------------------------------------
__global__ void pack_w_kernel(const float* __restrict__ W, _Float16* __restrict__ P, int K) {
  int t = blockIdx.x * blockDim.x + threadIdx.x;
  int total = K * 256;
  if (t >= total) return;
  int j    = t & 15;
  int lane = (t >> 4) & 31;
  int rest = t >> 9;
  int ct   = rest & 15;
  int kt   = rest >> 4;
  int col  = ct * 16 + (lane & 15);
  int k    = kt * 32 + ((lane >> 4) << 4) + j;
  P[t] = (_Float16)W[k * 256 + col];
}

// ---------------------------------------------------------------------------
// GEMM: Y[N,256] = X[N,K] @ W[K,256] + bias   (K in {64,128}, N % 16 == 0)
// One wave per (16-row tile, 64-col group): 4 WMMA accumulators.
// ---------------------------------------------------------------------------
__global__ __launch_bounds__(128) void gemm_f16_kernel(
    const float* __restrict__ X, int K,
    const _Float16* __restrict__ Wp, const float* __restrict__ bias,
    float* __restrict__ Y) {
  const int lane = threadIdx.x & 31;
  const int wv   = threadIdx.x >> 5;        // 0..3 -> 64-col group
  const int lrow = lane & 15;
  const int hi   = lane >> 4;               // lane half (selects K sub-rows)
  const size_t row = (size_t)blockIdx.x * 16 + lrow;

  v8f acc[4] = {};
  const int ktiles = K >> 5;
  for (int kt = 0; kt < ktiles; ++kt) {
    AFrag a;
    const float* xp = X + row * (size_t)K + kt * 32 + hi * 16;
#pragma unroll
    for (int j = 0; j < 8; ++j) {
      a.h[j]     = (_Float16)xp[j];       // K = kt*32 + hi*16 + j
      a.h[8 + j] = (_Float16)xp[8 + j];   // K = kt*32 + hi*16 + 8 + j
    }
    const _Float16* wb = Wp + (((size_t)kt * 16 + wv * 4) * 32 + lane) * 16;
#pragma unroll
    for (int t = 0; t < 4; ++t) {
      v16h b = *(const v16h*)(wb + (size_t)t * 32 * 16);
      acc[t] = __builtin_amdgcn_wmma_f32_16x16x32_f16(
          /*neg_a=*/false, a.v, /*neg_b=*/false, b,
          /*c_mod=*/(short)0, acc[t], /*reuse_a=*/false, /*reuse_b=*/false);
    }
  }
#pragma unroll
  for (int t = 0; t < 4; ++t) {
    int col = (wv * 4 + t) * 16 + lrow;
    float bv = bias[col];
#pragma unroll
    for (int j = 0; j < 8; ++j) {
      Y[((size_t)blockIdx.x * 16 + hi * 8 + j) * 256 + col] = acc[t][j] + bv;
    }
  }
}

// ---------------------------------------------------------------------------
// Edge pass A: logits[e,h] = sum_c leaky(xl[s,h,c]+xr[d,h,c]) * att[h,c]
// + segment-max into mEnc (monotonic uint encoding). One wave per (edge,head).
// ---------------------------------------------------------------------------
__global__ __launch_bounds__(256) void edge_logits_kernel(
    const float* __restrict__ xl, const float* __restrict__ xr,
    const float* __restrict__ att, const int* __restrict__ src,
    const int* __restrict__ dst, float* __restrict__ logit,
    unsigned* __restrict__ mEnc, int ngroups) {
  int g = blockIdx.x * 8 + (threadIdx.x >> 5);
  if (g >= ngroups) return;
  int e = g >> 1, h = g & 1;
  int lane = threadIdx.x & 31;
  int s = src[e], d = dst[e];
  const float* pl = xl + (size_t)s * 256 + h * 128;
  const float* pr = xr + (size_t)d * 256 + h * 128;
  const float* pa = att + h * 128;
  float sum = 0.f;
#pragma unroll
  for (int c = 0; c < 128; c += 32) {
    float v = pl[c + lane] + pr[c + lane];
    v = v > 0.f ? v : 0.2f * v;     // LeakyReLU(0.2)
    sum += v * pa[c + lane];
  }
  for (int off = 16; off > 0; off >>= 1) sum += __shfl_down(sum, off);
  if (lane == 0) {
    logit[g] = sum;
    unsigned u = __builtin_bit_cast(unsigned, sum);
    unsigned key = u ^ ((unsigned)(((int)u) >> 31) | 0x80000000u);
    atomicMax(&mEnc[(size_t)d * 2 + h], key);
  }
}

// ---------------------------------------------------------------------------
// Edge pass B: a = exp(logit - m[dst]); denom[dst] += a  (1 thread/(edge,head))
// ---------------------------------------------------------------------------
__global__ void edge_exp_kernel(const int* __restrict__ dst,
                                const unsigned* __restrict__ mEnc,
                                float* __restrict__ logit,
                                float* __restrict__ denom, int n) {
  int g = blockIdx.x * blockDim.x + threadIdx.x;
  if (g >= n) return;
  int e = g >> 1, h = g & 1;
  int d = dst[e];
  unsigned em = mEnc[(size_t)d * 2 + h];
  unsigned ub = (em & 0x80000000u) ? (em ^ 0x80000000u) : ~em;
  float m = __builtin_bit_cast(float, ub);
  float a = __expf(logit[g] - m);
  logit[g] = a;
  atomicAddF(&denom[(size_t)d * 2 + h], a);
}

// ---------------------------------------------------------------------------
// Edge pass C: acc[dst,h,:] += xl[src,h,:] * (a / (denom+eps)). Wave/(edge,head).
// ---------------------------------------------------------------------------
__global__ __launch_bounds__(256) void edge_aggr_kernel(
    const float* __restrict__ xl, const int* __restrict__ src,
    const int* __restrict__ dst, const float* __restrict__ logit,
    const float* __restrict__ denom, float* __restrict__ acc, int ngroups) {
  int g = blockIdx.x * 8 + (threadIdx.x >> 5);
  if (g >= ngroups) return;
  int e = g >> 1, h = g & 1;
  int lane = threadIdx.x & 31;
  int s = src[e], d = dst[e];
  float alpha = logit[g] / (denom[(size_t)d * 2 + h] + 1e-16f);
  const float* p = xl + (size_t)s * 256 + h * 128;
  float* q = acc + (size_t)d * 256 + h * 128;
#pragma unroll
  for (int c = 0; c < 128; c += 32)
    atomicAddF(&q[c + lane], p[c + lane] * alpha);
}

// ---------------------------------------------------------------------------
// Head-mean + bias + ReLU -> next-layer features [N,128]
// ---------------------------------------------------------------------------
__global__ void finalize_kernel(const float* __restrict__ acc,
                                const float* __restrict__ bias,
                                float* __restrict__ hout, long long n) {
  long long t = (long long)blockIdx.x * blockDim.x + threadIdx.x;
  if (t >= n) return;
  long long node = t >> 7;
  int c = (int)(t & 127);
  float v = 0.5f * (acc[node * 256 + c] + acc[node * 256 + 128 + c]) + bias[c];
  hout[t] = v > 0.f ? v : 0.f;
}

// ---------------------------------------------------------------------------
// Root selection: segment_min(i where x[i,0]==0) over batch
// ---------------------------------------------------------------------------
__global__ void root_find_kernel(const float* __restrict__ x,
                                 const int* __restrict__ batch,
                                 int* __restrict__ root, int N, int INF0) {
  int i = blockIdx.x * blockDim.x + threadIdx.x;
  if (i >= N) return;
  if (x[(size_t)i * INF0] == 0.0f) atomicMin(&root[batch[i]], i);
}

// ---------------------------------------------------------------------------
// out[b,o] = h[root[b],:] @ fcW[:,o] + fcb[o]
// ---------------------------------------------------------------------------
__global__ void out_kernel(const float* __restrict__ h, const float* __restrict__ W,
                           const float* __restrict__ b, const int* __restrict__ root,
                           float* __restrict__ out, int n, int N) {
  int t = blockIdx.x * blockDim.x + threadIdx.x;
  if (t >= n) return;
  int g = t >> 4, o = t & 15;
  int r = root[g];
  if (r >= N || r < 0) r = 0;
  float s = b[o];
  const float* hp = h + (size_t)r * 128;
  for (int k = 0; k < 128; ++k) s += hp[k] * W[k * 16 + o];
  out[t] = s;
}

// ---------------------------------------------------------------------------
extern "C" void kernel_launch(void* const* d_in, const int* in_sizes, int n_in,
                              void* d_out, int out_size, void* d_ws, size_t ws_size,
                              hipStream_t stream) {
  const float* x    = (const float*)d_in[0];
  const int* esrc   = (const int*)d_in[1];
  const int* edst   = (const int*)d_in[2];
  const int* batch  = (const int*)d_in[3];
  const int N    = in_sizes[3];
  const int Etot = in_sizes[1];
  const int IN0  = in_sizes[0] / N;          // 64
  const int Kl[3] = { IN0, 128, 128 };

  const float *Wl[3], *bl[3], *Wr[3], *br[3], *att[3], *bias[3];
  for (int l = 0; l < 3; ++l) {
    Wl[l]   = (const float*)d_in[4 + l * 6 + 0];
    bl[l]   = (const float*)d_in[4 + l * 6 + 1];
    Wr[l]   = (const float*)d_in[4 + l * 6 + 2];
    br[l]   = (const float*)d_in[4 + l * 6 + 3];
    att[l]  = (const float*)d_in[4 + l * 6 + 4];
    bias[l] = (const float*)d_in[4 + l * 6 + 5];
  }
  const float* fcW = (const float*)d_in[22];
  const float* fcb = (const float*)d_in[23];

  // ---- workspace layout (float units, 32B-aligned chunks) ----
  float* f = (float*)d_ws;
  size_t nf = 0;
  float* xl   = f + nf; nf += (size_t)N * 256;
  float* xr   = f + nf; nf += (size_t)N * 256;   // reused as `acc` after logits
  float* hbuf = f + nf; nf += (size_t)N * 128;
  float* logit = f + nf; nf += (((size_t)Etot * 2 + 7) & ~(size_t)7);
  float* denom = f + nf; nf += (((size_t)N * 2 + 7) & ~(size_t)7);
  unsigned* mEnc = (unsigned*)(f + nf); nf += (((size_t)N * 2 + 7) & ~(size_t)7);
  int* root = (int*)(f + nf); nf += 64;
  _Float16* wpack = (_Float16*)(f + nf);
  _Float16 *wlp[3], *wrp[3];
  size_t wo = 0;
  for (int l = 0; l < 3; ++l) {
    wlp[l] = wpack + wo; wo += (size_t)Kl[l] * 256;
    wrp[l] = wpack + wo; wo += (size_t)Kl[l] * 256;
  }

  // ---- pack weights into WMMA B-fragment layout (tiny, once per call) ----
  for (int l = 0; l < 3; ++l) {
    int n = Kl[l] * 256;
    pack_w_kernel<<<(n + 255) / 256, 256, 0, stream>>>(Wl[l], wlp[l], Kl[l]);
    pack_w_kernel<<<(n + 255) / 256, 256, 0, stream>>>(Wr[l], wrp[l], Kl[l]);
  }

  const int rowTiles = N / 16;
  const int ng = Etot * 2;
  const float* Xin = x;
  for (int l = 0; l < 3; ++l) {
    gemm_f16_kernel<<<rowTiles, 128, 0, stream>>>(Xin, Kl[l], wlp[l], bl[l], xl);
    gemm_f16_kernel<<<rowTiles, 128, 0, stream>>>(Xin, Kl[l], wrp[l], br[l], xr);
    fill_u32_kernel<<<(N * 2 + 255) / 256, 256, 0, stream>>>(mEnc, 0u, (long long)N * 2);
    fill_u32_kernel<<<(N * 2 + 255) / 256, 256, 0, stream>>>((unsigned*)denom, 0u, (long long)N * 2);
    edge_logits_kernel<<<(ng + 7) / 8, 256, 0, stream>>>(xl, xr, att[l], esrc, edst,
                                                         logit, mEnc, ng);
    // xr is dead now; zero it and reuse as the aggregation accumulator
    fill_u32_kernel<<<(int)(((long long)N * 256 + 255) / 256), 256, 0, stream>>>(
        (unsigned*)xr, 0u, (long long)N * 256);
    edge_exp_kernel<<<(ng + 255) / 256, 256, 0, stream>>>(edst, mEnc, logit, denom, ng);
    edge_aggr_kernel<<<(ng + 7) / 8, 256, 0, stream>>>(xl, esrc, edst, logit, denom, xr, ng);
    finalize_kernel<<<(int)(((long long)N * 128 + 255) / 256), 256, 0, stream>>>(
        xr, bias[l], hbuf, (long long)N * 128);
    Xin = hbuf;
  }

  fill_u32_kernel<<<1, 64, 0, stream>>>((unsigned*)root, (unsigned)N, 64);
  root_find_kernel<<<(N + 255) / 256, 256, 0, stream>>>(x, batch, root, N, IN0);
  out_kernel<<<(out_size + 255) / 256, 256, 0, stream>>>(hbuf, fcW, fcb, root,
                                                         (float*)d_out, out_size, N);
}
